// PixtralAttention_38628935860508
// MI455X (gfx1250) — matrix operand order repaired and checked by reading, not compiled, not run on hardware
//
#include <hip/hip_runtime.h>
#include <hip/hip_bf16.h>

#define SQ 3136
#define EE 1024
#define HH 16
#define DD 64
#define SB 196      /* SQ/16 */
#define KC32 98     /* SQ/32 */

typedef __attribute__((ext_vector_type(16))) __bf16 v16bf;
typedef __attribute__((ext_vector_type(8)))  float  v8f;
typedef __attribute__((ext_vector_type(4)))  float  nf4;   // native vec for NT stores

union Frag {
    v16bf v;
    unsigned short h[16];
    unsigned int   u[8];
    uint4          q[2];
};

__device__ __forceinline__ unsigned short f2bf(float f) {
    return __builtin_bit_cast(unsigned short, static_cast<__bf16>(f));
}

// CDNA5 async global->LDS DMA (ASYNCcnt-tracked). VDST = raw LDS byte offset
// (low 32 bits of the flat shared address per the aperture mapping).
__device__ __forceinline__ void async_ld_b128(unsigned lds_off, const float* g) {
    asm volatile("global_load_async_to_lds_b128 %0, %1, off"
                 :: "v"(lds_off), "v"(g) : "memory");
}
__device__ __forceinline__ void wait_async0() {
    asm volatile("s_wait_asynccnt 0x0" ::: "memory");
}

// stage a 128(e) x 32(k) fp32 weight tile into LDS via async DMA
__device__ __forceinline__ void stage_tile_async(const float* __restrict__ w,
                                                 int e0, int kk, unsigned ldsbase,
                                                 int tid) {
    #pragma unroll
    for (int it = 0; it < 4; ++it) {
        int i  = tid + it * 256;      // float4 granule index, 1024 total
        int r  = i >> 3;
        int c4 = i & 7;
        async_ld_b128(ldsbase + (unsigned)((r * 32 + c4 * 4) * 4),
                      w + (size_t)(e0 + r) * EE + kk + c4 * 4);
    }
}

// ---------------------------------------------------------------------------
// Kernel 1: QKV projection (hidden @ W^T) + RoPE, emit bf16 fragment-ready
// layouts (exact CDNA5 WMMA per-lane packing) for the attention kernel.
//   qh: A-frag  [H][SB][2 dchunk][32 lane][16 j]
//   kh: B-frag  [H][SB][2 dchunk][32 lane][16 j]   (K^T: k=d, n=key)
//   vh: B-frag  [H][KC32][4 dchunk][32 lane][16 j] (k=key, n=d)
// Weight tiles double-buffered in LDS via async DMA.
// ---------------------------------------------------------------------------
__global__ __launch_bounds__(256)
void qkv_rope_kernel(const float* __restrict__ hidden,
                     const float* __restrict__ cosb,
                     const float* __restrict__ sinb,
                     const float* __restrict__ wq,
                     const float* __restrict__ wk,
                     const float* __restrict__ wv,
                     unsigned short* __restrict__ qh,
                     unsigned short* __restrict__ kh,
                     unsigned short* __restrict__ vh)
{
    extern __shared__ __align__(16) char smem1[];
    unsigned short* lds_hidden = (unsigned short*)smem1;            // 32 KB bf16
    float* wbuf0   = (float*)(smem1 + 32768);                       // 16 KB fp32
    float* wbuf1   = (float*)(smem1 + 32768 + 16384);               // 16 KB fp32
    float* lds_out = (float*)(smem1 + 32768 + 32768);               // 8 KB

    const int tid  = threadIdx.x;
    const int lane = tid & 31;
    const int wid  = tid >> 5;
    const int sb   = blockIdx.x;       // query-row block (0..195)
    const int s0   = sb * 16;
    const int e0   = blockIdx.y * 128; // output-feature block
    const int z    = blockIdx.z;       // 0=q 1=k 2=v
    const float* w = (z == 0) ? wq : (z == 1) ? wk : wv;

    // stage 16 hidden rows as bf16 into LDS (read once, reused for 32 k-steps)
    for (int i = tid; i < 16 * 256; i += 256) {       // float4 granules
        int r  = i >> 8;
        int c4 = i & 255;
        const float4 f = ((const float4*)(hidden + (size_t)(s0 + r) * EE))[c4];
        int base = r * 1024 + c4 * 4;
        lds_hidden[base + 0] = f2bf(f.x);
        lds_hidden[base + 1] = f2bf(f.y);
        lds_hidden[base + 2] = f2bf(f.z);
        lds_hidden[base + 3] = f2bf(f.w);
    }

    const unsigned int* h32 = (const unsigned int*)lds_hidden;
    const unsigned wlds0 = (unsigned)(size_t)wbuf0;
    const unsigned wlds1 = (unsigned)(size_t)wbuf1;
    const int eloc  = wid * 16;
    const int mrow  = lane & 15;
    const int ahalf = (lane & 16) >> 1;   // +8 k for upper half-wave (unused; A frag uses bhalf)
    const int bhalf = lane & 16;          // +16 k for upper half-wave (B frag)

    // prime the pipeline: tile 0 -> wbuf0
    stage_tile_async(w, e0, 0, wlds0, tid);

    v8f acc = {};
    for (int t = 0; t < 32; ++t) {
        const float* wt = (t & 1) ? wbuf1 : wbuf0;
        wait_async0();          // own async loads (incl. tile t) complete
        __syncthreads();        // all waves' loads + previous reads complete
        if (t < 31)             // overlap DMA of tile t+1 with compute of t
            stage_tile_async(w, e0, (t + 1) * 32, (t & 1) ? wlds0 : wlds1, tid);

        const int kk = t * 32;
        Frag a, b;
        #pragma unroll
        for (int p = 0; p < 8; ++p) {
            int ka = 2 * p + bhalf;   // A k-pair base
            a.u[p] = h32[(mrow * 1024 + kk + ka) >> 1];
            float2 f2 = *(const float2*)(wt + (eloc + mrow) * 32 + 2 * p + bhalf);
            b.h[2 * p]     = f2bf(f2.x);
            b.h[2 * p + 1] = f2bf(f2.y);
        }
        acc = __builtin_amdgcn_wmma_f32_16x16x32_bf16(
                  false, a.v, false, b.v, (short)0, acc, false, false);
    }

    __syncthreads();
    // spill C tile to LDS so RoPE can pair d <-> d+-32 across wave tiles
    #pragma unroll
    for (int r = 0; r < 8; ++r) {
        int m = r + ((lane & 16) >> 1);
        int n = lane & 15;
        lds_out[m * 128 + eloc + n] = acc[r];
    }
    __syncthreads();

    for (int i = tid; i < 16 * 128; i += 256) {
        int m = i >> 7;
        int c = i & 127;
        int e = e0 + c;
        int h = e >> 6;
        int d = e & 63;
        int s = s0 + m;
        float x = lds_out[i];
        float val;
        if (z < 2) {
            float xr = (d < 32) ? -lds_out[m * 128 + (c + 32)]
                                :  lds_out[m * 128 + (c - 32)];
            val = x * cosb[s * 64 + d] + xr * sinb[s * 64 + d];
        } else {
            val = x;
        }
        unsigned short bv = f2bf(val);
        if (z == 0) {            // Q -> A-frag layout
            int k  = d & 31, dc = d >> 5;
            int ln = (s & 15) + (((k >> 3) & 1) << 4);
            int j  = (k & 7) + ((k & 16) >> 1);
            qh[((((size_t)h * SB + sb) * 2 + dc) * 32 + ln) * 16 + j] = bv;
        } else if (z == 1) {     // K -> B-frag (k=d, n=key)
            int kc = d & 31, dc = d >> 5;
            int ln = (s & 15) + (((kc >> 3) & 1) << 4);
            int j  = (kc & 7) + ((kc & 16) >> 1);
            kh[((((size_t)h * SB + sb) * 2 + dc) * 32 + ln) * 16 + j] = bv;
        } else {                 // V -> B-frag (k=key, n=d)
            int kc  = s & 31;
            int kcb = s >> 5;
            int dc  = d >> 4;
            int ln  = (d & 15) + (kc & 16);
            int j   = kc & 15;
            vh[((((size_t)h * KC32 + kcb) * 4 + dc) * 32 + ln) * 16 + j] = bv;
        }
    }
}

// ---------------------------------------------------------------------------
// Kernel 2: attention for one (head, 16-query block).
// Full 16 x 3136 score block held in LDS (~197 KB dynamic LDS).
// attn_weights streamed to HBM with non-temporal float4 stores.
// ---------------------------------------------------------------------------
__global__ __launch_bounds__(256)
void attn_kernel(const unsigned short* __restrict__ qh,
                 const unsigned short* __restrict__ kh,
                 const unsigned short* __restrict__ vh,
                 float* __restrict__ attn_w,          // [H][S][S]
                 unsigned short* __restrict__ aout)   // A-frag [SB][32][32][16]
{
    extern __shared__ __align__(16) char smem_raw[];
    float* sc  = (float*)smem_raw;                         // [16][3136]
    float* red = (float*)(smem_raw + (size_t)16 * SQ * 4); // [16][16]
    float* inv = red + 256;                                // [16]

    const int tid = threadIdx.x, lane = tid & 31, wid = tid >> 5;
    const int qb = blockIdx.x;
    const int h  = blockIdx.y;
    const float scale = 0.125f;   // 1/sqrt(64)

    // Q A-fragments for this query block (broadcast across all 8 waves)
    Frag qa0, qa1;
    {
        const uint4* p0 = (const uint4*)(qh + ((((size_t)h * SB + qb) * 2 + 0) * 32 + lane) * 16);
        const uint4* p1 = (const uint4*)(qh + ((((size_t)h * SB + qb) * 2 + 1) * 32 + lane) * 16);
        qa0.q[0] = p0[0]; qa0.q[1] = p0[1];
        qa1.q[0] = p1[0]; qa1.q[1] = p1[1];
    }

    // Phase A: scores = scale * Q K^T, key blocks round-robined over waves
    for (int kb = wid; kb < SB; kb += 8) {
        const uint4* p0 = (const uint4*)(kh + ((((size_t)h * SB + kb) * 2 + 0) * 32 + lane) * 16);
        const uint4* p1 = (const uint4*)(kh + ((((size_t)h * SB + kb) * 2 + 1) * 32 + lane) * 16);
        if (kb + 8 < SB) {   // prefetch next block's fragments under the WMMAs
            __builtin_prefetch(kh + ((((size_t)h * SB + kb + 8) * 2 + 0) * 32 + lane) * 16, 0, 0);
            __builtin_prefetch(kh + ((((size_t)h * SB + kb + 8) * 2 + 1) * 32 + lane) * 16, 0, 0);
        }
        Frag b0, b1;
        b0.q[0] = p0[0]; b0.q[1] = p0[1];
        b1.q[0] = p1[0]; b1.q[1] = p1[1];
        v8f c = {};
        c = __builtin_amdgcn_wmma_f32_16x16x32_bf16(false, qa0.v, false, b0.v, (short)0, c, false, false);
        c = __builtin_amdgcn_wmma_f32_16x16x32_bf16(false, qa1.v, false, b1.v, (short)0, c, false, false);
        #pragma unroll
        for (int r = 0; r < 8; ++r) {
            int m = r + ((lane & 16) >> 1);
            int n = lane & 15;
            sc[m * SQ + kb * 16 + n] = c[r] * scale;
        }
    }
    __syncthreads();

    // Phase B: softmax, 16 threads per query row, float4 passes
    const int row = tid >> 4, sub = tid & 15;
    nf4* scr = (nf4*)(sc + row * SQ);           // 784 vec4 per row
    float mx = -3.4e38f;
    for (int i = sub; i < 784; i += 16) {
        nf4 f = scr[i];
        mx = fmaxf(mx, fmaxf(fmaxf(f.x, f.y), fmaxf(f.z, f.w)));
    }
    red[row * 16 + sub] = mx;
    __syncthreads();
    float rmax = -3.4e38f;
    #pragma unroll
    for (int t = 0; t < 16; ++t) rmax = fmaxf(rmax, red[row * 16 + t]);
    __syncthreads();
    float sum = 0.f;
    for (int i = sub; i < 784; i += 16) {
        nf4 f = scr[i];
        f.x = __expf(f.x - rmax);
        f.y = __expf(f.y - rmax);
        f.z = __expf(f.z - rmax);
        f.w = __expf(f.w - rmax);
        scr[i] = f;                      // keep unnormalized exp in LDS
        sum += (f.x + f.y) + (f.z + f.w);
    }
    red[row * 16 + sub] = sum;
    __syncthreads();
    float rsum = 0.f;
    #pragma unroll
    for (int t = 0; t < 16; ++t) rsum += red[row * 16 + t];
    float rinv = 1.0f / rsum;
    if (sub == 0) inv[row] = rinv;
    {   // stream normalized attn_weights to HBM: dominant traffic, write-once
        nf4* dst4 = (nf4*)(attn_w + ((size_t)h * SQ + (size_t)(qb * 16 + row)) * SQ);
        for (int i = sub; i < 784; i += 16) {
            nf4 f = scr[i] * rinv;
            __builtin_nontemporal_store(f, dst4 + i);   // NT: don't pollute L2
        }
    }
    __syncthreads();

    // Phase C: O = P V, key 32-chunks round-robined over waves
    v8f o0 = {}, o1 = {}, o2 = {}, o3 = {};
    const int m = lane & 15;
    for (int kc = wid; kc < KC32; kc += 8) {
        const unsigned short* vb = vh + ((size_t)h * KC32 + kc) * 4 * 512;
        if (kc + 8 < KC32)
            __builtin_prefetch(vh + ((size_t)h * KC32 + kc + 8) * 4 * 512 + lane * 16, 0, 0);
        Frag a;
        #pragma unroll
        for (int j = 0; j < 16; ++j) {
            int k = kc * 32 + j + (lane & 16);
            a.h[j] = f2bf(sc[m * SQ + k]);
        }
        #pragma unroll
        for (int dc = 0; dc < 4; ++dc) {
            Frag b;
            const uint4* p = (const uint4*)(vb + (size_t)dc * 512 + lane * 16);
            b.q[0] = p[0]; b.q[1] = p[1];
            v8f& o = (dc == 0) ? o0 : (dc == 1) ? o1 : (dc == 2) ? o2 : o3;
            o = __builtin_amdgcn_wmma_f32_16x16x32_bf16(false, a.v, false, b.v, (short)0, o, false, false);
        }
    }
    __syncthreads();                 // scores dead; reuse LDS for reduction
    float* ored = sc;                // [8 waves][16 m][64 d]
    #pragma unroll
    for (int dc = 0; dc < 4; ++dc) {
        v8f o = (dc == 0) ? o0 : (dc == 1) ? o1 : (dc == 2) ? o2 : o3;
        #pragma unroll
        for (int r = 0; r < 8; ++r) {
            int mm = r + ((lane & 16) >> 1);
            int dd = dc * 16 + (lane & 15);
            ored[(wid * 16 + mm) * 64 + dd] = o[r];
        }
    }
    __syncthreads();
    for (int i = tid; i < 1024; i += 256) {
        int mm = i >> 6, dd = i & 63;
        float ssum = 0.f;
        #pragma unroll
        for (int wv2 = 0; wv2 < 8; ++wv2) ssum += ored[(wv2 * 16 + mm) * 64 + dd];
        ssum *= inv[mm];                       // fold in softmax 1/sum here
        int e  = h * 64 + dd;
        int ec = e >> 5, k = e & 31;
        int ln = mm + (((k >> 3) & 1) << 4);
        int j  = (k & 7) + ((k & 16) >> 1);
        aout[(((size_t)qb * 32 + ec) * 32 + ln) * 16 + j] = f2bf(ssum);
    }
}

// ---------------------------------------------------------------------------
// Kernel 3: out = attn_out @ wo^T (fp32 result), async double-buffered weights
// ---------------------------------------------------------------------------
__global__ __launch_bounds__(256)
void outproj_kernel(const unsigned short* __restrict__ aout,
                    const float* __restrict__ wo,
                    float* __restrict__ out)
{
    __shared__ __align__(16) float wbuf[2][128 * 32];   // 32 KB fp32
    const int tid = threadIdx.x, lane = tid & 31, wid = tid >> 5;
    const int sb = blockIdx.x;
    const int e0 = blockIdx.y * 128;
    const int eloc  = wid * 16;
    const int n     = lane & 15;
    const int bhalf = lane & 16;
    const unsigned wlds0 = (unsigned)(size_t)&wbuf[0][0];
    const unsigned wlds1 = (unsigned)(size_t)&wbuf[1][0];

    stage_tile_async(wo, e0, 0, wlds0, tid);

    v8f acc = {};
    for (int t = 0; t < 32; ++t) {
        const float* wt = wbuf[t & 1];
        wait_async0();
        __syncthreads();
        if (t < 31)
            stage_tile_async(wo, e0, (t + 1) * 32, (t & 1) ? wlds0 : wlds1, tid);

        Frag a, b;
        const uint4* pa = (const uint4*)(aout + (((size_t)sb * 32 + t) * 32 + lane) * 16);
        a.q[0] = pa[0]; a.q[1] = pa[1];
        #pragma unroll
        for (int p = 0; p < 8; ++p) {
            float2 f2 = *(const float2*)(wt + (eloc + n) * 32 + (2 * p < 8 ? 2 * p : 2 * p + 8) + ((lane & 16) >> 1));
            b.h[2 * p]     = f2bf(f2.x);
            b.h[2 * p + 1] = f2bf(f2.y);
        }
        acc = __builtin_amdgcn_wmma_f32_16x16x32_bf16(
                  false, a.v, false, b.v, (short)0, acc, false, false);
    }
    #pragma unroll
    for (int r = 0; r < 8; ++r) {
        int m = r + ((lane & 16) >> 1);
        out[(size_t)(sb * 16 + m) * EE + e0 + eloc + n] = acc[r];
    }
}

// ---------------------------------------------------------------------------
extern "C" void kernel_launch(void* const* d_in, const int* in_sizes, int n_in,
                              void* d_out, int out_size, void* d_ws, size_t ws_size,
                              hipStream_t stream) {
    (void)in_sizes; (void)n_in; (void)out_size; (void)ws_size;
    const float* hidden = (const float*)d_in[0];
    const float* cosb   = (const float*)d_in[1];
    const float* sinb   = (const float*)d_in[2];
    const float* wq     = (const float*)d_in[3];
    const float* wk     = (const float*)d_in[4];
    const float* wv     = (const float*)d_in[5];
    const float* wo     = (const float*)d_in[6];

    float* out    = (float*)d_out;                 // [S][E]
    float* attn_w = out + (size_t)SQ * EE;         // [H][S][S]

    const size_t fragElems = (size_t)SQ * EE;      // 3,211,264 bf16 each
    unsigned short* qh   = (unsigned short*)d_ws;
    unsigned short* kh   = qh + fragElems;
    unsigned short* vh   = kh + fragElems;
    unsigned short* aout = vh + fragElems;

    const size_t smem1 = 32768 + 16384 + 16384 + 8192;   // 72 KB
    (void)hipFuncSetAttribute((const void*)qkv_rope_kernel,
                              hipFuncAttributeMaxDynamicSharedMemorySize, (int)smem1);
    dim3 g1(SB, 8, 3);
    qkv_rope_kernel<<<g1, 256, smem1, stream>>>(hidden, cosb, sinb, wq, wk, wv, qh, kh, vh);

    const size_t smem2 = (size_t)16 * SQ * 4 + 256 * 4 + 16 * 4;  // ~197 KB
    (void)hipFuncSetAttribute((const void*)attn_kernel,
                              hipFuncAttributeMaxDynamicSharedMemorySize, (int)smem2);
    dim3 g2(SB, HH);
    attn_kernel<<<g2, 256, smem2, stream>>>(qh, kh, vh, attn_w, aout);

    dim3 g3(SB, 8);
    outproj_kernel<<<g3, 256, 0, stream>>>(aout, wo, out);
}
